// MoE_56014963474965
// MI455X (gfx1250) — compile-verified, matrix operand order checked by reading, not run on hardware
//
#include <hip/hip_runtime.h>
#include <hip/hip_bf16.h>

// ---- problem constants (match reference) ----
#define NE   8      // experts
#define TOPK 2
#define DM   1024   // d_model
#define FF   4096   // d_ff
#define NTOK 2048   // B*S
#define NSLOT (NTOK*TOPK)
#define EPSV 1e-6f

// ---- GEMM tiling ----
#define BM 64
#define BN 128
#define BK 32
#define MT (NTOK/BM)   // max M tiles per expert = 32

typedef __attribute__((ext_vector_type(16))) __bf16 v16bf;
typedef __attribute__((ext_vector_type(8)))  float  v8f;

union Frag { v16bf v; unsigned int u[8]; };

// ---- workspace layout (bytes) ----
#define OFF_COUNTS   0u          // 8 ints
#define OFF_COUNTS2  32u         // 8 ints
#define OFF_BASES    64u         // 8 ints
#define OFF_TOKID    128u        // NSLOT ints
#define OFF_TOKW     (OFF_TOKID + NSLOT*4u)
#define OFF_MU       (OFF_TOKW  + NSLOT*4u)
#define OFF_RS       (OFF_MU    + NTOK*4u)
#define OFF_IDX      (OFF_RS    + NTOK*4u)
#define OFF_WTOP     (OFF_IDX   + NSLOT*4u)
#define OFF_H        ((OFF_WTOP + NSLOT*4u + 255u) & ~255u)   // NSLOT*FF bf16

// CDNA5 async global->LDS copy (GLOBAL_LOAD_ASYNC_TO_LDS_B32, tracked by ASYNCcnt).
// lds: byte offset within LDS (low 32 bits of a generic pointer to __shared__,
// since the LDS aperture lives in addr[63:32] per ISA 10.2).
__device__ __forceinline__ void async_g2lds_b32(unsigned lds, const void* gaddr) {
    asm volatile("global_load_async_to_lds_b32 %0, %1, off"
                 :: "v"(lds), "v"(gaddr) : "memory");
}
__device__ __forceinline__ void wait_asynccnt0() {
    asm volatile("s_wait_asynccnt 0x0" ::: "memory");
}

// ================= gating: scores, top-2 softmax, LN stats =================
__global__ __launch_bounds__(256) void gate_kernel(
    const float* __restrict__ x, const float* __restrict__ Wg,
    float* __restrict__ muA, float* __restrict__ rsA,
    int* __restrict__ idxTop, float* __restrict__ wTop, int* __restrict__ counts)
{
    const int wave = threadIdx.x >> 5;
    const int lane = threadIdx.x & 31;
    const int t = blockIdx.x * 8 + wave;
    if (t >= NTOK) return;
    const float* xr = x + (size_t)t * DM;

    float s[NE];
#pragma unroll
    for (int e = 0; e < NE; ++e) s[e] = 0.f;
    float sum = 0.f, sq = 0.f;
    for (int d = lane; d < DM; d += 32) {
        float xv = xr[d];
        sum += xv; sq += xv * xv;
        const float* wr = Wg + (size_t)d * NE;
#pragma unroll
        for (int e = 0; e < NE; ++e) s[e] += xv * wr[e];
    }
#pragma unroll
    for (int off = 16; off > 0; off >>= 1) {
        sum += __shfl_xor(sum, off, 32);
        sq  += __shfl_xor(sq,  off, 32);
#pragma unroll
        for (int e = 0; e < NE; ++e) s[e] += __shfl_xor(s[e], off, 32);
    }
    if (lane == 0) {
        float mu = sum / (float)DM;
        float var = sq / (float)DM - mu * mu;
        muA[t] = mu;
        rsA[t] = rsqrtf(var + EPSV);
        int i0 = 0; float b0 = s[0];
#pragma unroll
        for (int e = 1; e < NE; ++e) if (s[e] > b0) { b0 = s[e]; i0 = e; }
        int i1 = -1; float b1v = -3.0e38f;
#pragma unroll
        for (int e = 0; e < NE; ++e) if (e != i0 && s[e] > b1v) { b1v = s[e]; i1 = e; }
        float e1 = __expf(b1v - b0);
        float inv = 1.f / (1.f + e1);
        idxTop[2 * t + 0] = i0; idxTop[2 * t + 1] = i1;
        wTop[2 * t + 0] = inv;  wTop[2 * t + 1] = e1 * inv;
        atomicAdd(&counts[i0], 1);
        atomicAdd(&counts[i1], 1);
    }
}

// ================= tiny scan over 8 experts =================
__global__ void scan_kernel(const int* __restrict__ counts, int* __restrict__ bases)
{
    if (threadIdx.x == 0) {
        int acc = 0;
        for (int e = 0; e < NE; ++e) { bases[e] = acc; acc += counts[e]; }
    }
}

// ================= scatter tokens into per-expert slot lists =================
__global__ __launch_bounds__(256) void scatter_kernel(
    const int* __restrict__ idxTop, const float* __restrict__ wTop,
    const int* __restrict__ bases, int* __restrict__ counts2,
    int* __restrict__ tokId, float* __restrict__ tokW)
{
    int t = blockIdx.x * blockDim.x + threadIdx.x;
    if (t >= NTOK) return;
#pragma unroll
    for (int k = 0; k < TOPK; ++k) {
        int e = idxTop[2 * t + k];
        int pos = atomicAdd(&counts2[e], 1);
        int slot = bases[e] + pos;
        tokId[slot] = t;
        tokW[slot] = wTop[2 * t + k];
    }
}

// ================= pass 1: H = relu(LN(x)*g+b @ W1 + b1), bf16 WMMA =================
__global__ __launch_bounds__(256) void ffn1_kernel(
    const float* __restrict__ x, const float* __restrict__ gamma, const float* __restrict__ beta,
    const float* __restrict__ W1, const float* __restrict__ b1,
    const int* __restrict__ tokId, const float* __restrict__ muA, const float* __restrict__ rsA,
    const int* __restrict__ counts, const int* __restrict__ bases,
    __bf16* __restrict__ H)
{
    const int e = blockIdx.z;
    const int mtile = blockIdx.y;
    const int cnt = counts[e];
    if (mtile * BM >= cnt) return;
    const int base = bases[e];
    const int nBase = blockIdx.x * BN;

    __shared__ __bf16 sA[BM][BK + 2];
    __shared__ __bf16 sB[BN][BK + 2];
    __shared__ int   rTok[BM];
    __shared__ float rMu[BM], rRs[BM];

    const int tid = threadIdx.x;
    if (tid < BM) {
        int sr = mtile * BM + tid;
        if (sr < cnt) {
            int tk = tokId[base + sr];
            rTok[tid] = tk; rMu[tid] = muA[tk]; rRs[tid] = rsA[tk];
        } else { rTok[tid] = -1; rMu[tid] = 0.f; rRs[tid] = 0.f; }
    }
    __syncthreads();

    const int wave = tid >> 5, lane = tid & 31;
    const int wm = wave >> 2, wn = wave & 3;
    const int half = lane >> 4, l16 = lane & 15;

    v8f acc[2][2] = {};

    const float* gRow = gamma + (size_t)e * DM;
    const float* bRow = beta  + (size_t)e * DM;

    for (int kt = 0; kt < DM; kt += BK) {
        // stage A: LN + per-expert affine, f32 -> bf16
        for (int i = tid; i < BM * BK; i += 256) {
            int m = i >> 5, k = i & (BK - 1);
            int tk = rTok[m];
            float v = 0.f;
            if (tk >= 0) {
                float xv = x[(size_t)tk * DM + kt + k];
                v = (xv - rMu[m]) * rRs[m] * gRow[kt + k] + bRow[kt + k];
            }
            sA[m][k] = (__bf16)v;
        }
        // stage B: W1 strip (K-major load, store n-major for pair gathers)
        for (int i = tid; i < BN * BK; i += 256) {
            int n = i & (BN - 1), k = i >> 7;
            float wv = W1[((size_t)e * DM + kt + k) * FF + nBase + n];
            sB[n][k] = (__bf16)wv;
        }
        if (kt + BK < DM)
            __builtin_prefetch(&W1[((size_t)e * DM + kt + BK) * FF + nBase + (tid & 127)], 0, 0);
        __syncthreads();

        Frag a[2], b[2];
#pragma unroll
        for (int mi = 0; mi < 2; ++mi) {
            int m = wm * 32 + mi * 16 + l16;
#pragma unroll
            for (int p = 0; p < 8; ++p) {
                int kk = ((p >> 2) << 4) + (half << 3) + ((p << 1) & 7);
                a[mi].u[p] = *(const unsigned int*)&sA[m][kk];
            }
        }
#pragma unroll
        for (int ni = 0; ni < 2; ++ni) {
            int n = wn * 32 + ni * 16 + l16;
#pragma unroll
            for (int p = 0; p < 8; ++p) {
                int kk = ((p >> 2) << 4) + (half << 3) + ((p << 1) & 7);
                b[ni].u[p] = *(const unsigned int*)&sB[n][kk];
            }
        }
#pragma unroll
        for (int mi = 0; mi < 2; ++mi)
#pragma unroll
            for (int ni = 0; ni < 2; ++ni)
                acc[mi][ni] = __builtin_amdgcn_wmma_f32_16x16x32_bf16(
                    false, a[mi].v, false, b[ni].v, (short)0, acc[mi][ni], false, false);
        __syncthreads();
    }

    // epilogue: +b1, relu, bf16 store to H
#pragma unroll
    for (int mi = 0; mi < 2; ++mi)
#pragma unroll
        for (int ni = 0; ni < 2; ++ni)
#pragma unroll
            for (int r = 0; r < 8; ++r) {
                int sr = mtile * BM + wm * 32 + mi * 16 + half * 8 + r;
                int col = nBase + wn * 32 + ni * 16 + l16;
                if (sr < cnt) {
                    float h = acc[mi][ni][r] + b1[(size_t)e * FF + col];
                    H[(size_t)(base + sr) * FF + col] = (__bf16)fmaxf(h, 0.f);
                }
            }
}

// ================= pass 2: y += w * (H @ W2 + b2), bf16 WMMA =================
__global__ __launch_bounds__(256) void ffn2_kernel(
    const __bf16* __restrict__ H, const float* __restrict__ W2, const float* __restrict__ b2,
    const int* __restrict__ tokId, const float* __restrict__ tokW,
    const int* __restrict__ counts, const int* __restrict__ bases,
    float* __restrict__ y)
{
    const int e = blockIdx.z;
    const int mtile = blockIdx.y;
    const int cnt = counts[e];
    if (mtile * BM >= cnt) return;
    const int base = bases[e];
    const int nBase = blockIdx.x * BN;

    __shared__ __bf16 sA[BM][BK + 2];
    __shared__ __bf16 sB[BN][BK + 2];

    const int tid = threadIdx.x;
    const int wave = tid >> 5, lane = tid & 31;
    const int wm = wave >> 2, wn = wave & 3;
    const int half = lane >> 4, l16 = lane & 15;

    v8f acc[2][2] = {};

    for (int kt = 0; kt < FF; kt += BK) {
        // stage A: H tile is already bf16 in global -> pure byte copy via
        // CDNA5 async global->LDS DMA (no VGPR round trip, ASYNCcnt tracked).
        for (int i = tid; i < (BM * BK) / 2; i += 256) {
            int m = i >> 4;            // 16 b32 words per 32-elem row
            int kw = (i & 15) << 1;    // bf16 column (even)
            int sr = mtile * BM + m;
            unsigned lds = (unsigned)(unsigned long long)&sA[m][kw];
            if (sr < cnt) {
                const void* g = (const void*)&H[(size_t)(base + sr) * FF + kt + kw];
                async_g2lds_b32(lds, g);
            } else {
                *(unsigned int*)&sA[m][kw] = 0u;   // zero-fill tail rows
            }
        }
        // stage B: W2 strip needs f32 -> bf16 conversion, keep register path
        for (int i = tid; i < BN * BK; i += 256) {
            int n = i & (BN - 1), k = i >> 7;
            float wv = W2[((size_t)e * FF + kt + k) * DM + nBase + n];
            sB[n][k] = (__bf16)wv;
        }
        if (kt + BK < FF)
            __builtin_prefetch(&W2[((size_t)e * FF + kt + BK) * DM + nBase + (tid & 127)], 0, 0);
        wait_asynccnt0();
        __syncthreads();

        Frag a[2], b[2];
#pragma unroll
        for (int mi = 0; mi < 2; ++mi) {
            int m = wm * 32 + mi * 16 + l16;
#pragma unroll
            for (int p = 0; p < 8; ++p) {
                int kk = ((p >> 2) << 4) + (half << 3) + ((p << 1) & 7);
                a[mi].u[p] = *(const unsigned int*)&sA[m][kk];
            }
        }
#pragma unroll
        for (int ni = 0; ni < 2; ++ni) {
            int n = wn * 32 + ni * 16 + l16;
#pragma unroll
            for (int p = 0; p < 8; ++p) {
                int kk = ((p >> 2) << 4) + (half << 3) + ((p << 1) & 7);
                b[ni].u[p] = *(const unsigned int*)&sB[n][kk];
            }
        }
#pragma unroll
        for (int mi = 0; mi < 2; ++mi)
#pragma unroll
            for (int ni = 0; ni < 2; ++ni)
                acc[mi][ni] = __builtin_amdgcn_wmma_f32_16x16x32_bf16(
                    false, a[mi].v, false, b[ni].v, (short)0, acc[mi][ni], false, false);
        __syncthreads();
    }

    // epilogue: weighted atomic accumulate into y (y was pre-initialized to x)
#pragma unroll
    for (int mi = 0; mi < 2; ++mi)
#pragma unroll
        for (int ni = 0; ni < 2; ++ni)
#pragma unroll
            for (int r = 0; r < 8; ++r) {
                int sr = mtile * BM + wm * 32 + mi * 16 + half * 8 + r;
                int col = nBase + wn * 32 + ni * 16 + l16;
                if (sr < cnt) {
                    int t = tokId[base + sr];
                    float w = tokW[base + sr];
                    float o = acc[mi][ni][r] + b2[(size_t)e * DM + col];
                    atomicAdd(&y[(size_t)t * DM + col], w * o);
                }
            }
}

extern "C" void kernel_launch(void* const* d_in, const int* in_sizes, int n_in,
                              void* d_out, int out_size, void* d_ws, size_t ws_size,
                              hipStream_t stream) {
    const float* x     = (const float*)d_in[0];
    const float* Wg    = (const float*)d_in[1];
    const float* gamma = (const float*)d_in[2];
    const float* beta  = (const float*)d_in[3];
    const float* W1    = (const float*)d_in[4];
    const float* b1    = (const float*)d_in[5];
    const float* W2    = (const float*)d_in[6];
    const float* b2    = (const float*)d_in[7];
    float* y = (float*)d_out;

    char* ws = (char*)d_ws;
    int*    counts  = (int*)(ws + OFF_COUNTS);
    int*    counts2 = (int*)(ws + OFF_COUNTS2);
    int*    bases   = (int*)(ws + OFF_BASES);
    int*    tokId   = (int*)(ws + OFF_TOKID);
    float*  tokW    = (float*)(ws + OFF_TOKW);
    float*  muA     = (float*)(ws + OFF_MU);
    float*  rsA     = (float*)(ws + OFF_RS);
    int*    idxTop  = (int*)(ws + OFF_IDX);
    float*  wTop    = (float*)(ws + OFF_WTOP);
    __bf16* H       = (__bf16*)(ws + OFF_H);

    hipMemsetAsync(ws, 0, 128, stream);
    hipMemcpyAsync(y, x, (size_t)NTOK * DM * sizeof(float),
                   hipMemcpyDeviceToDevice, stream);

    gate_kernel<<<NTOK / 8, 256, 0, stream>>>(x, Wg, muA, rsA, idxTop, wTop, counts);
    scan_kernel<<<1, 32, 0, stream>>>(counts, bases);
    scatter_kernel<<<NTOK / 256, 256, 0, stream>>>(idxTop, wTop, bases, counts2, tokId, tokW);

    ffn1_kernel<<<dim3(FF / BN, MT, NE), 256, 0, stream>>>(
        x, gamma, beta, W1, b1, tokId, muA, rsA, counts, bases, H);
    ffn2_kernel<<<dim3(DM / BN, MT, NE), 256, 0, stream>>>(
        H, W2, b2, tokId, tokW, counts, bases, y);
}
